// LSTMTagger_80152679678457
// MI455X (gfx1250) — matrix operand order repaired and checked by reading, not compiled, not run on hardware
//
#include <hip/hip_runtime.h>

// ---------------------------------------------------------------------------
// LSTM tagger for gfx1250 (MI455X), all GEMMs via v_wmma_f32_16x16x32_f16.
// ---------------------------------------------------------------------------

typedef __attribute__((ext_vector_type(16))) _Float16 v16h;
typedef __attribute__((ext_vector_type(8)))  float    v8f;

#define B_    64
#define S_    256
#define C_    12
#define EC    50
#define HC    64
#define EW    300
#define HW    320
#define DIN   428
#define DINP  448     // DIN padded to multiple of 32
#define NTAG  20
#define NWORDS (B_ * S_)   // 16384

#define WMMA_F16(a, b, c) \
  __builtin_amdgcn_wmma_f32_16x16x32_f16(false, (a), false, (b), (short)0, (c), false, false)

// Branch-free activations: v_exp_f32 + v_rcp_f32, no EXEC divergence.
// fast_tanh saturates via inf/0: x>>0 -> exp=inf -> rcp=0 -> 1; x<<0 -> -1.
static __device__ __forceinline__ float fast_sigmoid(float x) {
  return __builtin_amdgcn_rcpf(1.0f + __expf(-x));
}
static __device__ __forceinline__ float fast_tanh(float x) {
  return 1.0f - 2.0f * __builtin_amdgcn_rcpf(__expf(2.0f * x) + 1.0f);
}

// A fragment (16x32, row-major source, leading dim lda), CDNA5 16-bit A layout:
// lane = half*16 + m ; element e -> K = 16*(e/8) + 8*half + (e%8)
static __device__ __forceinline__ v16h load_a_frag(const _Float16* base, int lda, int lane) {
  const int hf = (lane >> 4) & 1;
  const int m  = lane & 15;
  const _Float16* r = base + m * lda;
  v16h a;
#pragma unroll
  for (int e = 0; e < 8; ++e) a[e] = r[hf * 8 + e];
#pragma unroll
  for (int e = 0; e < 8; ++e) a[8 + e] = r[16 + hf * 8 + e];
  return a;
}

// B fragment from pre-packed buffer: tile = nt*KT + kt, 512 f16 per tile,
// per-lane 16 contiguous f16 (32 bytes).
static __device__ __forceinline__ v16h load_b_frag(const _Float16* pack, int tile, int lane) {
  return *(const v16h*)(pack + (size_t)tile * 512 + lane * 16);
}

// ---------------------------------------------------------------------------
// Weight pack: src = W [Nrows x K] (row-major, torch layout), B-operand = W^T.
// Packed B layout per 32x16 tile: element (lane,e) -> k = kt*32 + 16*(e/8) + 8*(lane/16) + (e%8),
//                                                      n = nt*16 + (lane%16)
// ---------------------------------------------------------------------------
__global__ void pack_weight_kernel(const float* __restrict__ src, int Nrows, int K,
                                   int KT, int NT, _Float16* __restrict__ dst)
{
  int total = NT * KT * 512;
  for (int idx = blockIdx.x * 256 + threadIdx.x; idx < total; idx += gridDim.x * 256) {
    int tile = idx >> 9;
    int r    = idx & 511;
    int lane = r >> 4, e = r & 15;
    int nt = tile / KT, kt = tile - nt * KT;
    int hf = lane >> 4, nn = lane & 15;
    int k = kt * 32 + (e >> 3) * 16 + hf * 8 + (e & 7);
    int n = nt * 16 + nn;
    float v = (k < K && n < Nrows) ? src[(size_t)n * K + k] : 0.0f;
    dst[idx] = (_Float16)v;
  }
}

__global__ void add_vec_kernel(const float* __restrict__ a, const float* __restrict__ b,
                               float* __restrict__ d, int n)
{
  int i = blockIdx.x * 256 + threadIdx.x;
  if (i < n) d[i] = a[i] + b[i];
}

// ---------------------------------------------------------------------------
// Char BiLSTM: grid (256 word-blocks, 2 dirs), 256 threads (8 waves).
// Block owns 64 words; 12 serial steps; gates [64 x 256].
// Wave w: row-tile mt = w&3, column quadruple pair jh = w>>2 (2 j-tiles,
// each with i/f/g/o tiles). Input projection (K=50 padded 64) fused.
// ---------------------------------------------------------------------------
__global__ void __launch_bounds__(256)
char_lstm_kernel(const int* __restrict__ char_ids, const float* __restrict__ char_emb,
                 const float* __restrict__ h0, const float* __restrict__ c0,
                 const _Float16* __restrict__ pWihF, const _Float16* __restrict__ pWihB,
                 const _Float16* __restrict__ pWhhF, const _Float16* __restrict__ pWhhB,
                 const float* __restrict__ cbF, const float* __restrict__ cbB,
                 float* __restrict__ char_feat)
{
  __shared__ _Float16 ceA[64 * 64];   // input tile for current char position
  __shared__ _Float16 hA[64 * 64];    // hidden state (f16, A-operand source)
  __shared__ int      sCid[64 * C_];  // all char ids for this block's 64 words
  const int tid  = threadIdx.x;
  const int lane = tid & 31;
  const int wv   = tid >> 5;
  const int mt   = wv & 3;
  const int jh   = wv >> 2;           // 0..1
  const int hf   = (lane >> 4) & 1;
  const int nn   = lane & 15;
  const int blk  = blockIdx.x;
  const int dir  = blockIdx.y;
  const _Float16* pWih = dir ? pWihB : pWihF;
  const _Float16* pWhh = dir ? pWhhB : pWhhF;
  const float*    cb   = dir ? cbB   : cbF;

  for (int i = tid; i < 64 * C_; i += 256) sCid[i] = char_ids[blk * 64 * C_ + i];
  for (int i = tid; i < 64 * 64; i += 256) hA[i] = (_Float16)h0[dir * HC + (i & 63)];

  v8f cc[2];
#pragma unroll
  for (int jt = 0; jt < 2; ++jt) {
    float cv = c0[dir * HC + (jh * 2 + jt) * 16 + nn];
#pragma unroll
    for (int v = 0; v < 8; ++v) cc[jt][v] = cv;
  }
  __syncthreads();

  for (int t = 0; t < C_; ++t) {
    const int tc = dir ? (C_ - 1 - t) : t;
    for (int i = tid; i < 64 * 64; i += 256) {
      int m = i >> 6, k = i & 63;
      ceA[i] = (k < EC) ? (_Float16)char_emb[(size_t)sCid[m * C_ + tc] * EC + k]
                        : (_Float16)0.0f;
    }
    __syncthreads();

    v8f acc[2][4];
#pragma unroll
    for (int jt = 0; jt < 2; ++jt)
#pragma unroll
      for (int g = 0; g < 4; ++g) {
        float bv = cb[(g * 4 + jh * 2 + jt) * 16 + nn];
#pragma unroll
        for (int v = 0; v < 8; ++v) acc[jt][g][v] = bv;
      }
#pragma unroll
    for (int kt = 0; kt < 2; ++kt) {          // fused input projection
      v16h a = load_a_frag(ceA + mt * 16 * 64 + kt * 32, 64, lane);
#pragma unroll
      for (int jt = 0; jt < 2; ++jt)
#pragma unroll
        for (int g = 0; g < 4; ++g) {
          int nt = g * 4 + jh * 2 + jt;
          v16h b = load_b_frag(pWih, nt * 2 + kt, lane);
          acc[jt][g] = WMMA_F16(a, b, acc[jt][g]);
        }
    }
#pragma unroll
    for (int kt = 0; kt < 2; ++kt) {          // hidden recurrence
      v16h a = load_a_frag(hA + mt * 16 * 64 + kt * 32, 64, lane);
#pragma unroll
      for (int jt = 0; jt < 2; ++jt)
#pragma unroll
        for (int g = 0; g < 4; ++g) {
          int nt = g * 4 + jh * 2 + jt;
          v16h b = load_b_frag(pWhh, nt * 2 + kt, lane);
          acc[jt][g] = WMMA_F16(a, b, acc[jt][g]);
        }
    }
    __syncthreads();

#pragma unroll
    for (int jt = 0; jt < 2; ++jt) {
      int jc = (jh * 2 + jt) * 16 + nn;
#pragma unroll
      for (int v = 0; v < 8; ++v) {
        float iv = fast_sigmoid(acc[jt][0][v]);
        float fv = fast_sigmoid(acc[jt][1][v]);
        float gv = fast_tanh(acc[jt][2][v]);
        float ov = fast_sigmoid(acc[jt][3][v]);
        float cv = fv * cc[jt][v] + iv * gv;
        cc[jt][v] = cv;
        float hv = ov * fast_tanh(cv);
        int m = mt * 16 + hf * 8 + v;
        hA[m * 64 + jc] = (_Float16)hv;
        if (t == C_ - 1)
          char_feat[(size_t)(blk * 64 + m) * 128 + dir * 64 + jc] = hv;
      }
    }
    __syncthreads();
  }
}

// ---------------------------------------------------------------------------
// Build word-LSTM input x[s*64+b][448] f16 = [word_emb(300) | char fwd(64) |
// char bwd(64) | zero pad(20)]
// ---------------------------------------------------------------------------
__global__ void build_x_kernel(const int* __restrict__ sent, const float* __restrict__ word_emb,
                               const float* __restrict__ char_feat, _Float16* __restrict__ xA)
{
  const size_t total = (size_t)NWORDS * DINP;
  for (size_t idx = (size_t)blockIdx.x * 256 + threadIdx.x; idx < total;
       idx += (size_t)gridDim.x * 256) {
    int row = (int)(idx / DINP);
    int k   = (int)(idx - (size_t)row * DINP);
    int s = row >> 6, b = row & 63;
    float v = 0.0f;
    if (k < EW)       v = word_emb[(size_t)sent[b * S_ + s] * EW + k];
    else if (k < DIN) v = char_feat[(size_t)(b * S_ + s) * 128 + (k - EW)];
    xA[idx] = (_Float16)v;
  }
}

// ---------------------------------------------------------------------------
// Word input projection GEMM: [16384 x 448] x [448 x 2560] -> Pre (f16,
// C-fragment layout: tile ((s*4+btile)*160 + nt), 256 f16/tile, lane*8+v).
// Grid (256 s, 5 n-groups), 256 threads. A staged in LDS (57344 B dynamic).
// ---------------------------------------------------------------------------
__global__ void __launch_bounds__(256)
word_inproj_kernel(const _Float16* __restrict__ xA, const _Float16* __restrict__ pBw,
                   _Float16* __restrict__ Pre)
{
  extern __shared__ _Float16 sX[];   // 64 x 448
  const int tid = threadIdx.x, lane = tid & 31, wv = tid >> 5;
  const int mt = wv >> 1, nh = wv & 1;
  const int s = blockIdx.x;
  const int ngBase = blockIdx.y * 32 + nh * 16;
  for (int i = tid; i < 64 * DINP; i += 256) sX[i] = xA[(size_t)s * 64 * DINP + i];
  __syncthreads();

  v8f acc[16];
#pragma unroll
  for (int j = 0; j < 16; ++j)
#pragma unroll
    for (int v = 0; v < 8; ++v) acc[j][v] = 0.0f;

  for (int kt = 0; kt < 14; ++kt) {
    v16h a = load_a_frag(sX + mt * 16 * DINP + kt * 32, DINP, lane);
#pragma unroll
    for (int j = 0; j < 16; ++j) {
      v16h b = load_b_frag(pBw, (ngBase + j) * 14 + kt, lane);
      acc[j] = WMMA_F16(a, b, acc[j]);
    }
  }
#pragma unroll
  for (int j = 0; j < 16; ++j) {
    _Float16* p = Pre + ((size_t)(s * 4 + mt) * 160 + ngBase + j) * 256 + lane * 8;
#pragma unroll
    for (int v = 0; v < 8; ++v) p[v] = (_Float16)acc[j][v];
  }
}

// ---------------------------------------------------------------------------
// Word BiLSTM recurrence: grid (4 batch-tiles, 2 dirs), 640 threads (20 waves).
// Wave j owns column quadruple (i,f,g,o) at j-tile j; c stays in VGPRs,
// h ping-pongs through LDS (f16). 256 serial steps, 800 WMMA/step/block.
// Next step's Pre fragments are prefetched (global_prefetch_b8) during GEMM.
// ---------------------------------------------------------------------------
__global__ void __launch_bounds__(640)
word_lstm_kernel(const _Float16* __restrict__ Pre,
                 const _Float16* __restrict__ pWhhF, const _Float16* __restrict__ pWhhB,
                 const float* __restrict__ wbF, const float* __restrict__ wbB,
                 _Float16* __restrict__ xO)
{
  __shared__ _Float16 hA[16 * HW];
  const int tid = threadIdx.x, lane = tid & 31, j = tid >> 5;   // j = 0..19
  const int hf = (lane >> 4) & 1, nn = lane & 15;
  const int btile = blockIdx.x, dir = blockIdx.y;
  const _Float16* pW = dir ? pWhhB : pWhhF;
  const float*    wb = dir ? wbB   : wbF;

  // Loop-invariant gate biases, hoisted out of the 256-step loop.
  float bvs[4];
#pragma unroll
  for (int g = 0; g < 4; ++g) bvs[g] = wb[(20 * g + j) * 16 + nn];

  for (int i = tid; i < 16 * HW; i += 640) hA[i] = (_Float16)0.0f;
  v8f cc;
#pragma unroll
  for (int v = 0; v < 8; ++v) cc[v] = 0.0f;
  __syncthreads();

  for (int step = 0; step < S_; ++step) {
    const int sv = dir ? (S_ - 1 - step) : step;
    v8f acc[4];
#pragma unroll
    for (int g = 0; g < 4; ++g) {
      int nt = 20 * g + j;
      const _Float16* pf =
          Pre + (((size_t)(sv * 4 + btile) * 160) + dir * 80 + nt) * 256 + lane * 8;
#pragma unroll
      for (int v = 0; v < 8; ++v) acc[g][v] = bvs[g] + (float)pf[v];
    }
    // Prefetch next step's gate pre-activations while this step's GEMM runs.
    if (step + 1 < S_) {
      const int svn = dir ? (S_ - 2 - step) : (step + 1);
#pragma unroll
      for (int g = 0; g < 4; ++g) {
        const _Float16* pf =
            Pre + (((size_t)(svn * 4 + btile) * 160) + dir * 80 + (20 * g + j)) * 256 + lane * 8;
        __builtin_prefetch(pf, 0, 0);
      }
    }
    for (int kt = 0; kt < 10; ++kt) {
      v16h a = load_a_frag(hA + kt * 32, HW, lane);
#pragma unroll
      for (int g = 0; g < 4; ++g) {
        v16h b = load_b_frag(pW, (20 * g + j) * 10 + kt, lane);
        acc[g] = WMMA_F16(a, b, acc[g]);
      }
    }
    __syncthreads();
    const int jc = j * 16 + nn;
#pragma unroll
    for (int v = 0; v < 8; ++v) {
      float iv = fast_sigmoid(acc[0][v]);
      float fv = fast_sigmoid(acc[1][v]);
      float gv = fast_tanh(acc[2][v]);
      float ov = fast_sigmoid(acc[3][v]);
      float cv = fv * cc[v] + iv * gv;
      cc[v] = cv;
      float hv = ov * fast_tanh(cv);
      int m = hf * 8 + v;
      hA[m * HW + jc] = (_Float16)hv;
      xO[((size_t)sv * 64 + btile * 16 + m) * 640 + dir * HW + jc] = (_Float16)hv;
    }
    __syncthreads();
  }
}

// ---------------------------------------------------------------------------
// Tag GEMM: [16384 x 640] x [640 x 20(pad 32)] + b_tag -> out f32.
// Grid 512 blocks x 128 threads (4 waves, 32 rows/block). LDS 40960 B dynamic.
// ---------------------------------------------------------------------------
__global__ void __launch_bounds__(128)
tag_kernel(const _Float16* __restrict__ xO, const _Float16* __restrict__ pBt,
           const float* __restrict__ btag, float* __restrict__ out)
{
  extern __shared__ _Float16 sO[];   // 32 x 640
  const int tid = threadIdx.x, lane = tid & 31, wv = tid >> 5;
  const int mt = wv >> 1, nt = wv & 1;
  const int hf = (lane >> 4) & 1, nn = lane & 15;
  const int blk = blockIdx.x;
  for (int i = tid; i < 32 * 640; i += 128) sO[i] = xO[(size_t)blk * 32 * 640 + i];
  __syncthreads();

  const int ng = nt * 16 + nn;
  float bv = (ng < NTAG) ? btag[ng] : 0.0f;
  v8f acc;
#pragma unroll
  for (int v = 0; v < 8; ++v) acc[v] = bv;

  for (int kt = 0; kt < 20; ++kt) {
    v16h a = load_a_frag(sO + mt * 16 * 640 + kt * 32, 640, lane);
    v16h b = load_b_frag(pBt, nt * 20 + kt, lane);
    acc = WMMA_F16(a, b, acc);
  }
  if (ng < NTAG) {
#pragma unroll
    for (int v = 0; v < 8; ++v) {
      int m = mt * 16 + hf * 8 + v;
      out[(size_t)(blk * 32 + m) * NTAG + ng] = acc[v];
    }
  }
}

// ---------------------------------------------------------------------------
// Host orchestration
// ---------------------------------------------------------------------------
extern "C" void kernel_launch(void* const* d_in, const int* in_sizes, int n_in,
                              void* d_out, int out_size, void* d_ws, size_t ws_size,
                              hipStream_t stream)
{
  (void)in_sizes; (void)n_in; (void)out_size; (void)ws_size;
  const int*   sent     = (const int*)  d_in[0];
  const int*   char_ids = (const int*)  d_in[2];
  const float* word_emb = (const float*)d_in[3];
  const float* char_emb = (const float*)d_in[4];
  const float* char_h0  = (const float*)d_in[5];
  const float* char_c0  = (const float*)d_in[6];
  const float* Wih_cf = (const float*)d_in[7];
  const float* Whh_cf = (const float*)d_in[8];
  const float* bih_cf = (const float*)d_in[9];
  const float* bhh_cf = (const float*)d_in[10];
  const float* Wih_cb = (const float*)d_in[11];
  const float* Whh_cb = (const float*)d_in[12];
  const float* bih_cb = (const float*)d_in[13];
  const float* bhh_cb = (const float*)d_in[14];
  const float* Wih_wf = (const float*)d_in[15];
  const float* Whh_wf = (const float*)d_in[16];
  const float* bih_wf = (const float*)d_in[17];
  const float* bhh_wf = (const float*)d_in[18];
  const float* Wih_wb = (const float*)d_in[19];
  const float* Whh_wb = (const float*)d_in[20];
  const float* bih_wb = (const float*)d_in[21];
  const float* bhh_wb = (const float*)d_in[22];
  const float* W_tag  = (const float*)d_in[23];
  const float* b_tag  = (const float*)d_in[24];

  char* ws = (char*)d_ws;
  size_t off = 0;
  auto alloc = [&](size_t bytes) -> char* {
    char* p = ws + off;
    off += (bytes + 255) & ~(size_t)255;
    return p;
  };

  _Float16* pBcF   = (_Float16*)alloc((size_t)16 * 2 * 512 * 2);
  _Float16* pBcB   = (_Float16*)alloc((size_t)16 * 2 * 512 * 2);
  _Float16* pWhhCF = (_Float16*)alloc((size_t)16 * 2 * 512 * 2);
  _Float16* pWhhCB = (_Float16*)alloc((size_t)16 * 2 * 512 * 2);
  _Float16* pBwIn  = (_Float16*)alloc((size_t)160 * 14 * 512 * 2);
  _Float16* pWhhWF = (_Float16*)alloc((size_t)80 * 10 * 512 * 2);
  _Float16* pWhhWB = (_Float16*)alloc((size_t)80 * 10 * 512 * 2);
  _Float16* pBtag  = (_Float16*)alloc((size_t)2 * 20 * 512 * 2);
  float* cbF = (float*)alloc(256 * 4);
  float* cbB = (float*)alloc(256 * 4);
  float* wbF = (float*)alloc(1280 * 4);
  float* wbB = (float*)alloc(1280 * 4);
  float*    char_feat = (float*)   alloc((size_t)NWORDS * 128 * 4);
  _Float16* xA        = (_Float16*)alloc((size_t)NWORDS * DINP * 2);
  _Float16* Pre       = (_Float16*)alloc((size_t)S_ * 4 * 160 * 256 * 2);
  _Float16* xO        = (_Float16*)alloc((size_t)NWORDS * 640 * 2);

  auto packLaunch = [&](const float* src, int N, int K, int KT, int NT, _Float16* dst) {
    int total = NT * KT * 512;
    pack_weight_kernel<<<(total + 255) / 256, 256, 0, stream>>>(src, N, K, KT, NT, dst);
  };
  packLaunch(Wih_cf, 256, 50, 2, 16, pBcF);
  packLaunch(Wih_cb, 256, 50, 2, 16, pBcB);
  packLaunch(Whh_cf, 256, 64, 2, 16, pWhhCF);
  packLaunch(Whh_cb, 256, 64, 2, 16, pWhhCB);
  packLaunch(Wih_wf, 1280, 428, 14, 80, pBwIn);
  packLaunch(Wih_wb, 1280, 428, 14, 80, pBwIn + (size_t)80 * 14 * 512);
  packLaunch(Whh_wf, 1280, 320, 10, 80, pWhhWF);
  packLaunch(Whh_wb, 1280, 320, 10, 80, pWhhWB);
  packLaunch(W_tag, 20, 640, 20, 2, pBtag);

  add_vec_kernel<<<1, 256, 0, stream>>>(bih_cf, bhh_cf, cbF, 256);
  add_vec_kernel<<<1, 256, 0, stream>>>(bih_cb, bhh_cb, cbB, 256);
  add_vec_kernel<<<5, 256, 0, stream>>>(bih_wf, bhh_wf, wbF, 1280);
  add_vec_kernel<<<5, 256, 0, stream>>>(bih_wb, bhh_wb, wbB, 1280);

  char_lstm_kernel<<<dim3(NWORDS / 64, 2), 256, 0, stream>>>(
      char_ids, char_emb, char_h0, char_c0, pBcF, pBcB, pWhhCF, pWhhCB, cbF, cbB, char_feat);

  {
    size_t total = (size_t)NWORDS * DINP;
    int blocks = (int)((total + 255) / 256);
    build_x_kernel<<<blocks, 256, 0, stream>>>(sent, word_emb, char_feat, xA);
  }

  word_inproj_kernel<<<dim3(S_, 5), 256, 64 * DINP * 2, stream>>>(xA, pBwIn, Pre);

  word_lstm_kernel<<<dim3(4, 2), 640, 0, stream>>>(Pre, pWhhWF, pWhhWB, wbF, wbB, xO);

  tag_kernel<<<NWORDS / 32, 128, 32 * 640 * 2, stream>>>(xO, pBtag, b_tag, (float*)d_out);
}
